// FeatLSTM_32822140076679
// MI455X (gfx1250) — matrix operand order repaired and checked by reading, not compiled, not run on hardware
//
#include <hip/hip_runtime.h>
#include <hip/hip_bf16.h>
#include <math.h>

// Problem dims
#define B_  64
#define T_  512
#define D_  1024
#define H_  128      // layer-1 hidden
#define G1  512      // 4*H
#define O_  128      // layer-2 hidden

typedef __attribute__((ext_vector_type(16))) __bf16 v16bf;
typedef __attribute__((ext_vector_type(8)))  __bf16 v8bf;
typedef __attribute__((ext_vector_type(8)))  float  v8f;

static __device__ inline v8f v8zero() {
    v8f z;
#pragma unroll
    for (int i = 0; i < 8; ++i) z[i] = 0.0f;
    return z;
}

static __device__ inline float sigmoidf_(float x) {
    return 1.0f / (1.0f + __expf(-x));
}

// 16x32 bf16 A-fragment: lane holds row M=lane%16; lanes 0-15
// carry K={0..15}, lanes 16-31 K={16..31}. From a bf16 row base
// (already offset to [row][kbase]) that is two contiguous 16B loads.
static __device__ inline v16bf afrag(const __bf16* base, int khalf) {
    const v8bf lo = *(const v8bf*)(base + khalf * 16);
    const v8bf hi = *(const v8bf*)(base + khalf * 16 + 8);
    return __builtin_shufflevector(lo, hi, 0, 1, 2, 3, 4, 5, 6, 7,
                                   8, 9, 10, 11, 12, 13, 14, 15);
}

// ---------------------------------------------------------------------------
// Kernel 0: streaming f32 -> bf16 convert (feats and W_ih1), 4 elems/thread.
// ---------------------------------------------------------------------------
__global__ void __launch_bounds__(256)
tobf16(const float* __restrict__ in, __bf16* __restrict__ out) {
    const size_t i0 = ((size_t)blockIdx.x * 256 + threadIdx.x) * 4;
#pragma unroll
    for (int j = 0; j < 4; ++j) out[i0 + j] = (__bf16)in[i0 + j];
}

// ---------------------------------------------------------------------------
// Kernel 1: convert W_hh1 / W_ih2 / W_hh2 (each [512,128] f32) into
// WMMA-B-fragment-ordered bf16: frag[mat][ntile][kstep][lane][16 elems].
// B-layout (32x16): lane -> N = lane%16, K = (lane/16)*16 + e (contiguous),
// so each lane's fragment is ONE aligned 32-byte load in the scan.
// ---------------------------------------------------------------------------
__global__ void __launch_bounds__(32)
wfrag_convert(const float* __restrict__ Whh1, const float* __restrict__ Wih2,
              const float* __restrict__ Whh2, __bf16* __restrict__ wf) {
    const int blk  = blockIdx.x;        // 0..383 : 3 mats * 32 ntiles * 4 ksteps
    const int mat  = blk >> 7;
    const int fr   = blk & 127;
    const int nt   = fr >> 2;
    const int ks   = fr & 3;
    const int lane = threadIdx.x;
    const float* W = (mat == 0) ? Whh1 : (mat == 1) ? Wih2 : Whh2;
    const int n  = nt * 16 + (lane & 15);
    const int kb = ks * 32 + ((lane >> 4) << 4);
    __bf16* dst = wf + ((size_t)blk * 32 + lane) * 16;
#pragma unroll
    for (int e = 0; e < 16; ++e)
        dst[e] = (__bf16)W[(size_t)n * H_ + kb + e];
}

// ---------------------------------------------------------------------------
// Kernel 2: x1 = feats @ W_ih1^T + (b_ih1+b_hh1).  (32768x512)=(32768x1024)x
// (1024x512), bf16 WMMA, f32 accum.  LDS-free / barrier-free: A fragments are
// per-lane direct global loads (no cross-lane reuse); B fragments are 32B
// aligned global loads out of the L2-resident 1MB bf16 weight.
// Block = 256 thr (8 waves): wave does 2 M-strips x 4 N-tiles (block tile
// 256M x 64N), 8 WMMAs per 32-wide K step.
// ---------------------------------------------------------------------------
__global__ void __launch_bounds__(256)
gemm_x1(const __bf16* __restrict__ fb, const __bf16* __restrict__ wb,
        const float* __restrict__ bih1, const float* __restrict__ bhh1,
        float* __restrict__ x1) {
    const int tid   = threadIdx.x;
    const int lane  = tid & 31;
    const int w     = tid >> 5;
    const int mfr   = lane & 15;
    const int khalf = lane >> 4;
    const int Mbase = blockIdx.y * 256;
    const int Nbase = blockIdx.x * 64;

    const int m0 = Mbase + w * 32;       // first 16-row strip
    const int m1 = m0 + 16;              // second

    v8f acc[2][4];
#pragma unroll
    for (int mt = 0; mt < 2; ++mt)
#pragma unroll
        for (int nt = 0; nt < 4; ++nt) acc[mt][nt] = v8zero();

    const __bf16* arow0 = fb + (size_t)(m0 + mfr) * D_;
    const __bf16* arow1 = fb + (size_t)(m1 + mfr) * D_;

    for (int kb = 0; kb < D_; kb += 32) {
        const v16bf a0 = afrag(arow0 + kb, khalf);
        const v16bf a1 = afrag(arow1 + kb, khalf);
#pragma unroll
        for (int nt = 0; nt < 4; ++nt) {
            const int n = Nbase + nt * 16 + mfr;
            const v16bf b = *(const v16bf*)(wb + (size_t)n * D_ + kb + khalf * 16);
            acc[0][nt] = __builtin_amdgcn_wmma_f32_16x16x32_bf16(
                false, a0, false, b, (short)0, acc[0][nt], false, false);
            acc[1][nt] = __builtin_amdgcn_wmma_f32_16x16x32_bf16(
                false, a1, false, b, (short)0, acc[1][nt], false, false);
        }
    }

#pragma unroll
    for (int nt = 0; nt < 4; ++nt) {
        const int n   = Nbase + nt * 16 + mfr;
        const float bs = bih1[n] + bhh1[n];
#pragma unroll
        for (int mt = 0; mt < 2; ++mt) {
            const int mb = (mt ? m1 : m0) + (khalf << 3);
#pragma unroll
            for (int v = 0; v < 8; ++v)
                x1[(size_t)(mb + v) * G1 + n] = acc[mt][nt][v] + bs;
        }
    }
}

// ---------------------------------------------------------------------------
// Kernel 3: recurrent scan. 4 blocks x 256 thr (8 waves); block handles 16
// batch rows. Wave w owns gate columns j in [16w,16w+16) for ALL four gates
// (ntile = gate*8 + w), so LSTM gates + cell state stay in registers.
// h1/h2 live in LDS as ping-pong bf16 buffers -> 2 barriers per step.
// ---------------------------------------------------------------------------
__global__ void __launch_bounds__(256)
lstm_scan(const float* __restrict__ x1, const __bf16* __restrict__ wf,
          const float* __restrict__ bih2, const float* __restrict__ bhh2,
          float* __restrict__ out) {
    const int tid   = threadIdx.x;
    const int lane  = tid & 31;
    const int w     = tid >> 5;          // wave 0..7 == j-chunk
    const int mfr   = lane & 15;
    const int khalf = lane >> 4;
    const int bbase = blockIdx.x * 16;

    __shared__ __bf16 h1b[2][16 * H_];   // ping-pong, 4KB each
    __shared__ __bf16 h2b[2][16 * O_];

    for (int i = tid; i < 16 * H_; i += 256) {
        h1b[0][i] = (__bf16)0.0f; h1b[1][i] = (__bf16)0.0f;
        h2b[0][i] = (__bf16)0.0f; h2b[1][i] = (__bf16)0.0f;
    }
    __syncthreads();

    const __bf16* W1 = wf;                           // W_hh1 frags
    const __bf16* W2 = wf + (size_t)128 * 32 * 16;   // W_ih2 frags
    const __bf16* W3 = wf + (size_t)256 * 32 * 16;   // W_hh2 frags

    const int j = w * 16 + mfr;                      // this lane's column
    // layer-2 fused biases for this lane (constant over t and v)
    const float b2i = bih2[j]         + bhh2[j];
    const float b2f = bih2[128 + j]   + bhh2[128 + j];
    const float b2g = bih2[256 + j]   + bhh2[256 + j];
    const float b2o = bih2[384 + j]   + bhh2[384 + j];

    v8f c1 = v8zero(), c2 = v8zero();

    for (int t = 0; t < T_; ++t) {
        const int p = t & 1;

        // Prefetch next step's x1 slice (16 rows x 2KB = one 128B line/thread)
        {
            const int tp = (t + 1 < T_) ? t + 1 : t;
            const float* pf = x1 + ((size_t)(bbase + (tid >> 4)) * T_ + tp) * G1
                                 + (size_t)(tid & 15) * 32;
            __builtin_prefetch(pf, 0, 0);
        }

        // ---- GEMM1: gates = h1 @ W_hh1^T (register accs, one quartet/wave)
        v8f ai = v8zero(), af_ = v8zero(), ag = v8zero(), ao = v8zero();
#pragma unroll
        for (int ks = 0; ks < 4; ++ks) {
            const v16bf a = afrag(&h1b[p][mfr * H_ + ks * 32], khalf);
#pragma unroll
            for (int g = 0; g < 4; ++g) {
                const v16bf b = *(const v16bf*)&W1[(((size_t)(g * 8 + w) * 4 + ks) * 32 + lane) * 16];
                v8f* acc = (g == 0) ? &ai : (g == 1) ? &af_ : (g == 2) ? &ag : &ao;
                *acc = __builtin_amdgcn_wmma_f32_16x16x32_bf16(
                    false, a, false, b, (short)0, *acc, false, false);
            }
        }

        // ---- layer-1 gates in registers; write h1 into the OTHER buffer
#pragma unroll
        for (int v = 0; v < 8; ++v) {
            const int m = v + (khalf << 3);
            const size_t xr = ((size_t)(bbase + m) * T_ + t) * G1 + j;
            const float gi = ai[v]  + x1[xr];
            const float gf = af_[v] + x1[xr + 128];
            const float gg = ag[v]  + x1[xr + 256];
            const float go = ao[v]  + x1[xr + 384];
            const float cn = sigmoidf_(gf) * c1[v] + sigmoidf_(gi) * tanhf(gg);
            c1[v] = cn;
            h1b[p ^ 1][m * H_ + j] = (__bf16)(sigmoidf_(go) * tanhf(cn));
        }
        __syncthreads();   // all h1(t) visible

        // ---- GEMM2: gates = h1(t) @ W_ih2^T + h2(t-1) @ W_hh2^T
        ai = v8zero(); af_ = v8zero(); ag = v8zero(); ao = v8zero();
#pragma unroll
        for (int ks = 0; ks < 4; ++ks) {
            const v16bf a1 = afrag(&h1b[p ^ 1][mfr * H_ + ks * 32], khalf);
            const v16bf a2 = afrag(&h2b[p][mfr * O_ + ks * 32], khalf);
#pragma unroll
            for (int g = 0; g < 4; ++g) {
                const size_t fi = (((size_t)(g * 8 + w) * 4 + ks) * 32 + lane) * 16;
                const v16bf b2v = *(const v16bf*)&W2[fi];
                const v16bf b3v = *(const v16bf*)&W3[fi];
                v8f* acc = (g == 0) ? &ai : (g == 1) ? &af_ : (g == 2) ? &ag : &ao;
                *acc = __builtin_amdgcn_wmma_f32_16x16x32_bf16(
                    false, a1, false, b2v, (short)0, *acc, false, false);
                *acc = __builtin_amdgcn_wmma_f32_16x16x32_bf16(
                    false, a2, false, b3v, (short)0, *acc, false, false);
            }
        }

        // ---- layer-2 gates; write h2 ping-pong + streamed output
#pragma unroll
        for (int v = 0; v < 8; ++v) {
            const int m = v + (khalf << 3);
            const float gi = ai[v]  + b2i;
            const float gf = af_[v] + b2f;
            const float gg = ag[v]  + b2g;
            const float go = ao[v]  + b2o;
            const float cn = sigmoidf_(gf) * c2[v] + sigmoidf_(gi) * tanhf(gg);
            const float hn = sigmoidf_(go) * tanhf(cn);
            c2[v] = cn;
            h2b[p ^ 1][m * O_ + j] = (__bf16)hn;
            out[((size_t)(bbase + m) * T_ + t) * O_ + j] = hn;
        }
        __syncthreads();   // h2(t) visible; LDS reads of step t done
    }
}

// ---------------------------------------------------------------------------
extern "C" void kernel_launch(void* const* d_in, const int* in_sizes, int n_in,
                              void* d_out, int out_size, void* d_ws, size_t ws_size,
                              hipStream_t stream) {
    const float* feats = (const float*)d_in[0];
    const float* Wih1  = (const float*)d_in[1];
    const float* Whh1  = (const float*)d_in[2];
    const float* bih1  = (const float*)d_in[3];
    const float* bhh1  = (const float*)d_in[4];
    const float* Wih2  = (const float*)d_in[5];
    const float* Whh2  = (const float*)d_in[6];
    const float* bih2  = (const float*)d_in[7];
    const float* bhh2  = (const float*)d_in[8];
    float* out = (float*)d_out;

    // Workspace layout
    char* ws = (char*)d_ws;
    float*  x1     = (float*)ws;                                   // 67.1 MB
    ws += (size_t)B_ * T_ * G1 * sizeof(float);
    __bf16* wf     = (__bf16*)ws;                                  // 384 KB
    ws += (size_t)3 * 128 * 32 * 16 * sizeof(__bf16);
    __bf16* featsb = (__bf16*)ws;                                  // 67.1 MB
    ws += (size_t)B_ * T_ * D_ * sizeof(__bf16);
    __bf16* Wih1b  = (__bf16*)ws;                                  // 1 MB

    // f32 -> bf16 streaming converts
    tobf16<<<dim3(((size_t)B_ * T_ * D_) / 1024), dim3(256), 0, stream>>>(feats, featsb);
    tobf16<<<dim3(((size_t)G1 * D_) / 1024), dim3(256), 0, stream>>>(Wih1, Wih1b);

    // Recurrent weights -> WMMA fragment order (bf16)
    wfrag_convert<<<dim3(3 * 128), dim3(32), 0, stream>>>(Whh1, Wih2, Whh2, wf);

    // Big input GEMM
    dim3 g1(G1 / 64, (B_ * T_) / 256);   // (8, 128)
    gemm_x1<<<g1, dim3(256), 0, stream>>>(featsb, Wih1b, bih1, bhh1, x1);

    // Sequential scan, batch split across 4 independent workgroups
    lstm_scan<<<dim3(B_ / 16), dim3(256), 0, stream>>>(x1, wf, bih2, bhh2, out);
}
